// CustomBertSelfAttention_38732015075394
// MI455X (gfx1250) — matrix operand order repaired and checked by reading, not compiled, not run on hardware
//
#include <hip/hip_runtime.h>
#include <hip/hip_bf16.h>

// ---------------------------------------------------------------------------
// BERT self-attention forward for gfx1250 (MI455X), wave32 + WMMA.
//   B=16, S=512, H=1024, heads=16, head_size=64.
// Strategy: f16 operands / f32 accumulation through v_wmma_f32_16x16x32_f16.
// ---------------------------------------------------------------------------

typedef _Float16 h4   __attribute__((ext_vector_type(4)));
typedef _Float16 h8   __attribute__((ext_vector_type(8)));
typedef _Float16 v16h __attribute__((ext_vector_type(16)));
typedef float    v8f  __attribute__((ext_vector_type(8)));

union F16x16 { v16h v; h8 h[2]; };
union F32x8  { v8f  v; float4 f4[2]; };

#define WMMA16x16x32(a, b, c) \
  __builtin_amdgcn_wmma_f32_16x16x32_f16(false, (a), false, (b), (short)0, (c), false, false)

static constexpr int kB   = 16;
static constexpr int kS   = 512;
static constexpr int kH   = 1024;
static constexpr int kNH  = 16;
static constexpr int kHS  = 64;
static constexpr int kM   = kB * kS;          // 8192 rows for QKV GEMM
static constexpr float kLog2e = 1.4426950408889634f;

// ---------------------------------------------------------------------------
// Kernel 1: f32 -> f16 conversion (vectorized 4-wide)
// ---------------------------------------------------------------------------
__global__ __launch_bounds__(256) void cvt_f32_f16(const float* __restrict__ src,
                                                   _Float16* __restrict__ dst,
                                                   int n) {
  int i = (blockIdx.x * 256 + threadIdx.x) * 4;
  if (i < n) {
    float4 f = *(const float4*)(src + i);
    h4 o;
    o.x = (_Float16)f.x; o.y = (_Float16)f.y;
    o.z = (_Float16)f.z; o.w = (_Float16)f.w;
    *(h4*)(dst + i) = o;
  }
}

// ---------------------------------------------------------------------------
// Kernel 2: QKV projection GEMM (templated on output layout).
//   out[m,n] = sum_k x[m,k] * W[n,k] + bias[n]   (nn.Linear: x @ W^T + b)
//   x16 : [8192,1024] f16, W16 : [1024,1024] f16 row-major [o][h] (== B^T,
//   so B-fragments are two 16-byte loads from one W row per lane).
//   TRANSPOSE_V=false: Q,K stored [bh][s][hs] f16 (strided b16 stores).
//   TRANSPOSE_V=true : V stored [bh][hs][s] f16 (one packed b128 store/frag).
//   WG = 256 thr (8 waves, 2x4), wave tile 32x64 = 2x4 wmma accumulators:
//   8 WMMAs per 12 fragment loads per 32-wide K step.
// ---------------------------------------------------------------------------
template <bool TRANSPOSE_V>
__global__ __launch_bounds__(256) void qkv_gemm(const _Float16* __restrict__ x16,
                                                const _Float16* __restrict__ w16,
                                                const float* __restrict__ bias,
                                                _Float16* __restrict__ out) {
  const int lane = threadIdx.x & 31;
  const int wave = threadIdx.x >> 5;
  const int c    = lane & 15;      // row/col within 16-wide tile
  const int h    = lane >> 4;      // lane half
  const int mBase = blockIdx.x * 64  + (wave >> 2) * 32;
  const int nBase = blockIdx.y * 256 + (wave & 3) * 64;

  const _Float16* aRow0 = x16 + (size_t)(mBase + c) * kH + 8 * h;
  const _Float16* aRow1 = aRow0 + (size_t)16 * kH;

  v8f acc[2][4] = {};

  for (int k = 0; k < kH; k += 32) {
    F16x16 a[2], bf[4];
    // A 16x32 f16: lane -> row M = c ; K split lo/hi by lane half
    a[0].h[0] = *(const h8*)(aRow0 + k);         // K = 8h + 0..7
    a[0].h[1] = *(const h8*)(aRow0 + k + 16);    // K = 16 + 8h + 0..7
    a[1].h[0] = *(const h8*)(aRow1 + k);
    a[1].h[1] = *(const h8*)(aRow1 + k + 16);
    // prefetch the A stream a few K-steps ahead (global_prefetch_b8)
    __builtin_prefetch(aRow0 + k + 256, 0, 1);
    __builtin_prefetch(aRow1 + k + 256, 0, 1);
#pragma unroll
    for (int ni = 0; ni < 4; ++ni) {
      // B 32x16 f16: lane -> col N = c ; K = 8h + 0..7 and 16 + 8h + 0..7 in W row
      const _Float16* bp = w16 + (size_t)(nBase + ni * 16 + c) * kH + k + 8 * h;
      bf[ni].h[0] = *(const h8*)bp;
      bf[ni].h[1] = *(const h8*)(bp + 16);
    }
#pragma unroll
    for (int mi = 0; mi < 2; ++mi)
#pragma unroll
      for (int ni = 0; ni < 4; ++ni)
        acc[mi][ni] = WMMA16x16x32(a[mi].v, bf[ni].v, acc[mi][ni]);
  }

#pragma unroll
  for (int mi = 0; mi < 2; ++mi) {
    // D layout: lane(c,h), VGPR v -> M = mi*16 + 8h + v ; N = tile + c
    const int mRow = mBase + mi * 16 + 8 * h;    // first of 8 consecutive rows
    const int b    = mRow >> 9;                  // batch (tile never straddles)
    const int s0   = mRow & 511;                 // sequence base
#pragma unroll
    for (int ni = 0; ni < 4; ++ni) {
      const int n    = nBase + ni * 16 + c;
      const float bv = bias[n];
      const int head = n >> 6, d = n & 63;
      const size_t bh = (size_t)(b * kNH + head);
      if (TRANSPOSE_V) {
        // [bh][d][s]: v-elements contiguous along s -> one 16B store
        h8 pk;
#pragma unroll
        for (int v = 0; v < 8; ++v) pk[v] = (_Float16)(acc[mi][ni][v] + bv);
        *(h8*)(out + (bh * kHS + d) * kS + s0) = pk;
      } else {
        // [bh][s][d]: stride-64 b16 stores
        _Float16* op = out + (bh * kS + s0) * kHS + d;
#pragma unroll
        for (int v = 0; v < 8; ++v) op[v * kHS] = (_Float16)(acc[mi][ni][v] + bv);
      }
    }
  }
}

// ---------------------------------------------------------------------------
// Kernel 3: fused attention  softmax((Q K^T + mask)/8) V  per (b, head).
//   Scores computed transposed: S_t[key, query] = K x Q^T  so that softmax
//   state is per-lane (per query column). Probs go through a per-wave LDS
//   tile to become the B operand of  ctx^T[d, query] += V^T x P.
//   Grid: (256 bh, 4 query-slabs of 128). WG = 256 thr, wave owns 16 queries.
// ---------------------------------------------------------------------------
__global__ __launch_bounds__(256) void attn_fused(const _Float16* __restrict__ q16,
                                                  const _Float16* __restrict__ k16,
                                                  const _Float16* __restrict__ vt16,
                                                  const float* __restrict__ mask,
                                                  float* __restrict__ out) {
  __shared__ float mask_s[kS];
  __shared__ __align__(16) _Float16 probs_s[8][16][32];   // per-wave 1 KB tile

  const int bh = blockIdx.x;
  const int b = bh >> 4, head = bh & 15;

  for (int i = threadIdx.x; i < kS; i += 256) mask_s[i] = mask[b * kS + i];
  __syncthreads();

  const int lane = threadIdx.x & 31;
  const int wave = threadIdx.x >> 5;
  const int c = lane & 15, h = lane >> 4;
  const int qBase = blockIdx.y * 128 + wave * 16;

  // Q as B operand: B[K=d, N=query]; lane col = query c, K = 8h..8h+7 and 16+8h..16+8h+7
  const _Float16* qrow = q16 + ((size_t)bh * kS + qBase + c) * kHS;
  F16x16 qb[2];
  qb[0].h[0] = *(const h8*)(qrow + 8 * h);
  qb[0].h[1] = *(const h8*)(qrow + 16 + 8 * h);
  qb[1].h[0] = *(const h8*)(qrow + 32 + 8 * h);
  qb[1].h[1] = *(const h8*)(qrow + 48 + 8 * h);

  // loop-invariant row bases
  const _Float16* kBase0 = k16 + ((size_t)bh * kS + c) * kHS + 8 * h;
  const _Float16* vBase  = vt16 + ((size_t)bh * kHS + c) * kS + 8 * h;

  v8f ctx[4] = {};
  float M = -1e30f, l = 0.0f;

  for (int t = 0; t < kS; t += 32) {
    // ---- scores (transposed) for two 16-key blocks -----------------------
    v8f sc[2];
#pragma unroll
    for (int blk = 0; blk < 2; ++blk) {
      const _Float16* krow = kBase0 + (size_t)(t + blk * 16) * kHS;
      F16x16 ka0, ka1;                      // A[M=key, K=d]
      ka0.h[0] = *(const h8*)(krow);
      ka0.h[1] = *(const h8*)(krow + 16);
      ka1.h[0] = *(const h8*)(krow + 32);
      ka1.h[1] = *(const h8*)(krow + 48);
      v8f z = {};
      z = WMMA16x16x32(ka0.v, qb[0].v, z);
      z = WMMA16x16x32(ka1.v, qb[1].v, z);
      sc[blk] = z;
    }

    // ---- online softmax over this 32-key step ---------------------------
    float z0[8], z1[8];
    float bm = -1e30f;
#pragma unroll
    for (int v = 0; v < 8; ++v) {
      z0[v] = (sc[0][v] + mask_s[t + 8 * h + v]) * 0.125f;
      z1[v] = (sc[1][v] + mask_s[t + 16 + 8 * h + v]) * 0.125f;
      bm = fmaxf(bm, fmaxf(z0[v], z1[v]));
    }
    bm = fmaxf(bm, __shfl_xor(bm, 16, 32));   // combine lane halves (same query)
    const float Mn = fmaxf(M, bm);
    const float f  = exp2f((M - Mn) * kLog2e);

    float ps = 0.0f;
    h8 p0, p1;
#pragma unroll
    for (int v = 0; v < 8; ++v) {
      float e0 = exp2f((z0[v] - Mn) * kLog2e);
      float e1 = exp2f((z1[v] - Mn) * kLog2e);
      ps += e0 + e1;
      p0[v] = (_Float16)e0;
      p1[v] = (_Float16)e1;
    }
    ps += __shfl_xor(ps, 16, 32);
    l = l * f + ps;
    M = Mn;
#pragma unroll
    for (int db = 0; db < 4; ++db) ctx[db] *= f;

    // ---- C-layout -> B-layout via per-wave LDS tile (same-wave ordering) -
    *(h8*)&probs_s[wave][c][8 * h]      = p0;   // keys t + 8h + 0..7
    *(h8*)&probs_s[wave][c][16 + 8 * h] = p1;   // keys t + 16 + 8h + 0..7
    F16x16 pb;                                  // B[K=key(32), N=query]
    pb.h[0] = *(const h8*)&probs_s[wave][c][8 * h];
    pb.h[1] = *(const h8*)&probs_s[wave][c][16 + 8 * h];

    // ---- ctx^T[d, q] += V^T x P -----------------------------------------
#pragma unroll
    for (int db = 0; db < 4; ++db) {
      const _Float16* vrow = vBase + (size_t)(db * 16) * kS + t;
      F16x16 va;                                // A[M=d, K=key]
      va.h[0] = *(const h8*)(vrow);
      va.h[1] = *(const h8*)(vrow + 16);
      ctx[db] = WMMA16x16x32(va.v, pb.v, ctx[db]);
    }
  }

  // ---- finalize: divide by l, store f32 [B,S,H] --------------------------
  const float invl = 1.0f / l;
  const size_t rowBase = ((size_t)b * kS + qBase + c) * kH + head * kHS;
#pragma unroll
  for (int db = 0; db < 4; ++db) {
    F32x8 u;
    u.v = ctx[db] * invl;                       // lane: q=c, d=db*16+8h+v
    float* p = out + rowBase + db * 16 + 8 * h;
    *(float4*)(p)     = u.f4[0];
    *(float4*)(p + 4) = u.f4[1];
  }
}

// ---------------------------------------------------------------------------
// Host launcher
// ---------------------------------------------------------------------------
extern "C" void kernel_launch(void* const* d_in, const int* in_sizes, int n_in,
                              void* d_out, int out_size, void* d_ws, size_t ws_size,
                              hipStream_t stream) {
  const float* hidden = (const float*)d_in[0];
  const float* amask  = (const float*)d_in[1];
  const float* Wq     = (const float*)d_in[2];
  const float* bq     = (const float*)d_in[3];
  const float* Wk     = (const float*)d_in[4];
  const float* bk     = (const float*)d_in[5];
  const float* Wv     = (const float*)d_in[6];
  const float* bv     = (const float*)d_in[7];
  float* out = (float*)d_out;

  // Workspace carve-up (f16 elements): x16 | wq16 wk16 wv16 | q16 k16 vt16
  _Float16* x16  = (_Float16*)d_ws;
  _Float16* wq16 = x16  + (size_t)kM * kH;
  _Float16* wk16 = wq16 + (size_t)kH * kH;
  _Float16* wv16 = wk16 + (size_t)kH * kH;
  _Float16* q16  = wv16 + (size_t)kH * kH;
  _Float16* k16  = q16  + (size_t)kM * kH;
  _Float16* vt16 = k16  + (size_t)kM * kH;

  const int nX = kM * kH;   // 8,388,608
  const int nW = kH * kH;   // 1,048,576
  cvt_f32_f16<<<nX / 1024, 256, 0, stream>>>(hidden, x16, nX);
  cvt_f32_f16<<<nW / 1024, 256, 0, stream>>>(Wq, wq16, nW);
  cvt_f32_f16<<<nW / 1024, 256, 0, stream>>>(Wk, wk16, nW);
  cvt_f32_f16<<<nW / 1024, 256, 0, stream>>>(Wv, wv16, nW);

  dim3 gemmGrid(kM / 64, kH / 256);           // 128 x 4
  qkv_gemm<false><<<gemmGrid, 256, 0, stream>>>(x16, wq16, bq, q16);
  qkv_gemm<false><<<gemmGrid, 256, 0, stream>>>(x16, wk16, bk, k16);
  qkv_gemm<true ><<<gemmGrid, 256, 0, stream>>>(x16, wv16, bv, vt16);

  dim3 attnGrid(kB * kNH, kS / 128);          // 256 x 4
  attn_fused<<<attnGrid, 256, 0, stream>>>(q16, k16, vt16, amask, out);
}
